// EEGGraphConvNet_55937654063614
// MI455X (gfx1250) — compile-verified, matrix operand order checked by reading, not run on hardware
//
#include <hip/hip_runtime.h>
#include <hip/hip_bf16.h>

typedef __bf16 bf16_t;
typedef __attribute__((ext_vector_type(16))) __bf16 v16bf;
typedef __attribute__((ext_vector_type(8)))  __bf16 v8bf;
typedef __attribute__((ext_vector_type(8)))  float  v8f;
typedef __attribute__((ext_vector_type(4)))  int    v4i;

#define LRELU_SLOPE 0.01f
#define BN_EPS 1e-5f
#define N_GRAPHS 1024
#define POOL_DIM 256

// CDNA5 async global->LDS path (ASYNCcnt). Guarded so absence of the builtin
// falls back to the sync global_load + ds_store staging path.
#if defined(__has_builtin)
#  if __has_builtin(__builtin_amdgcn_global_load_async_to_lds_b128)
#    define USE_ASYNC_LDS 1
#  else
#    define USE_ASYNC_LDS 0
#  endif
#else
#  define USE_ASYNC_LDS 0
#endif

__device__ __forceinline__ float dleaky(float v) { return v > 0.f ? v : LRELU_SLOPE * v; }

#if USE_ASYNC_LDS
typedef __attribute__((address_space(1))) v4i gas_v4i;
typedef __attribute__((address_space(3))) v4i las_v4i;

// Copy 32 bytes (16 bf16) global -> LDS via two async B128 transfers.
// INST_OFFSET is added to BOTH the global and LDS addresses.
__device__ __forceinline__ void async_copy32(const bf16_t* g, bf16_t* l)
{
  gas_v4i* gp = (gas_v4i*)(void*)g;
  las_v4i* lp = (las_v4i*)(void*)l;
  __builtin_amdgcn_global_load_async_to_lds_b128(gp, lp, 0, 0);
  __builtin_amdgcn_global_load_async_to_lds_b128(gp, lp, 16, 0);
}
#endif

__device__ __forceinline__ void wait_async_all()
{
#if USE_ASYNC_LDS
#  if __has_builtin(__builtin_amdgcn_s_wait_asynccnt)
  __builtin_amdgcn_s_wait_asynccnt(0);
#  else
  asm volatile("s_wait_asynccnt 0x0" ::: "memory");
#  endif
#endif
}

// ---------------------------------------------------------------------------
// Conv stack: per node, conv1(1->8,k15,s2,p7) -> leaky -> conv2(8->16,k15,s2,p7)
// -> leaky -> conv3(16->1,k3,s1,p1). Output written as bf16, K padded 750->768.
// ---------------------------------------------------------------------------
__global__ __launch_bounds__(256) void conv_stack_kernel(
    const float* __restrict__ x,
    const float* __restrict__ w1, const float* __restrict__ b1,
    const float* __restrict__ w2, const float* __restrict__ b2,
    const float* __restrict__ w3, const float* __restrict__ b3,
    bf16_t* __restrict__ h0b)
{
  __shared__ float sw1[8 * 15];
  __shared__ float sb1[8];
  __shared__ float sw2[16 * 8 * 15];
  __shared__ float sb2[16];
  __shared__ float sw3[16 * 3];
  __shared__ float c2s[16][752];   // conv2 output with 1-col halo each side

  const int tid  = threadIdx.x;
  const int node = blockIdx.x;

  for (int i = tid; i < 8 * 15; i += 256)      sw1[i] = w1[i];
  for (int i = tid; i < 8; i += 256)           sb1[i] = b1[i];
  for (int i = tid; i < 16 * 8 * 15; i += 256) sw2[i] = w2[i];
  for (int i = tid; i < 16; i += 256)          sb2[i] = b2[i];
  for (int i = tid; i < 16 * 3; i += 256)      sw3[i] = w3[i];
  if (tid < 16) { c2s[tid][0] = 0.f; c2s[tid][751] = 0.f; }
  __syncthreads();

  const float* xr = x + (size_t)node * 3000;

  for (int p = tid; p < 750; p += 256) {
    float c1loc[15][8];
    #pragma unroll
    for (int t = 0; t < 15; ++t) {
      const int s = 2 * p - 7 + t;            // conv1 output position
      if (s < 0 || s >= 1500) {
        #pragma unroll
        for (int oc = 0; oc < 8; ++oc) c1loc[t][oc] = 0.f;
      } else {
        float xv[15];
        const int xbase = 2 * s - 7;
        #pragma unroll
        for (int k = 0; k < 15; ++k) {
          const int xi = xbase + k;
          xv[k] = (xi >= 0 && xi < 3000) ? xr[xi] : 0.f;
        }
        #pragma unroll
        for (int oc = 0; oc < 8; ++oc) {
          float a = sb1[oc];
          #pragma unroll
          for (int k = 0; k < 15; ++k) a += sw1[oc * 15 + k] * xv[k];
          c1loc[t][oc] = dleaky(a);
        }
      }
    }
    #pragma unroll
    for (int oc = 0; oc < 16; ++oc) {
      float a = sb2[oc];
      #pragma unroll
      for (int ic = 0; ic < 8; ++ic) {
        #pragma unroll
        for (int t = 0; t < 15; ++t)
          a += sw2[(oc * 8 + ic) * 15 + t] * c1loc[t][ic];
      }
      c2s[oc][p + 1] = dleaky(a);
    }
  }
  __syncthreads();

  const float bias3 = b3[0];
  bf16_t* out = h0b + (size_t)node * 768;
  for (int p = tid; p < 768; p += 256) {
    if (p < 750) {
      float a = bias3;
      #pragma unroll
      for (int ic = 0; ic < 16; ++ic) {
        #pragma unroll
        for (int k = 0; k < 3; ++k)
          a += sw3[ic * 3 + k] * c2s[ic][p + k];
      }
      out[p] = (bf16_t)a;
    } else {
      out[p] = (bf16_t)0.f;
    }
  }
}

// ---------------------------------------------------------------------------
// Weight conversion: W[K][N] fp32 row-major -> Wt[N][Kpad] bf16 (zero padded).
// ---------------------------------------------------------------------------
__global__ void wt_convert_kernel(const float* __restrict__ W, bf16_t* __restrict__ Wt,
                                  int K, int N, int Kpad)
{
  const size_t total = (size_t)N * (size_t)Kpad;
  for (size_t i = (size_t)blockIdx.x * blockDim.x + threadIdx.x; i < total;
       i += (size_t)gridDim.x * blockDim.x) {
    const int n = (int)(i / Kpad);
    const int k = (int)(i % Kpad);
    Wt[i] = (k < K) ? (bf16_t)W[(size_t)k * N + n] : (bf16_t)0.f;
  }
}

// ---------------------------------------------------------------------------
// WMMA bf16 GEMM: C[M][N] = A[M][K] @ Bt[N][K]^T (+bias). K multiple of 32.
// 128x128 block tile, BK=32, 8 waves each computing 64x32 (4x2 16x16 tiles).
// Double-buffered LDS; tiles staged with async global->LDS DMA (ASYNCcnt).
// ---------------------------------------------------------------------------
#define GBM 128
#define GBN 128
#define GBK 32

__device__ __forceinline__ v16bf load_frag_lds(const bf16_t* base, int row0, int lane)
{
  // 16-bit A/B fragment layout: lanes 0-15 hold K {0..7,16..23},
  // lanes 16-31 hold K {8..15,24..31}; row = row0 + (lane & 15).
  const int r  = row0 + (lane & 15);
  const int kh = (lane >> 4) * 8;
  const bf16_t* p = base + r * GBK + kh;
  v8bf lo = *(const v8bf*)(p);
  v8bf hi = *(const v8bf*)(p + 16);
  v16bf f;
  #pragma unroll
  for (int i = 0; i < 8; ++i) { f[i] = lo[i]; f[i + 8] = hi[i]; }
  return f;
}

__global__ __launch_bounds__(256) void gemm_bf16_kernel(
    const bf16_t* __restrict__ A,    // M x K
    const bf16_t* __restrict__ Bt,   // N x K (transposed weight)
    const float*  __restrict__ bias, // N or null
    float* __restrict__ Cf,          // M x N fp32 or null
    bf16_t* __restrict__ Cb,         // M x N bf16 or null
    int M, int N, int K)
{
  __shared__ __align__(16) bf16_t As[2][GBM][GBK];
  __shared__ __align__(16) bf16_t Bs[2][GBN][GBK];

  const int tid  = threadIdx.x;
  const int lane = tid & 31;
  const int wave = tid >> 5;       // 0..7
  const int wr   = wave >> 2;      // 0..1 : 64-row slab
  const int wc   = wave & 3;       // 0..3 : 32-col slab

  const int blockM = blockIdx.y * GBM;
  const int blockN = blockIdx.x * GBN;

  const v8f vzero = {0.f, 0.f, 0.f, 0.f, 0.f, 0.f, 0.f, 0.f};
  v8f acc[4][2];
  #pragma unroll
  for (int i = 0; i < 4; ++i)
    #pragma unroll
    for (int j = 0; j < 2; ++j) acc[i][j] = vzero;

  const int lr = tid >> 1;          // 0..127: tile row loaded by this thread
  const int lk = (tid & 1) * 16;    // 0 or 16: k-segment

  // Clamped staging rows: in this net M is a multiple of 128 and N of 128, so
  // the clamp never fires; if it did, garbage lands only in guarded-out rows.
  const int garow = (blockM + lr < M) ? (blockM + lr) : (M - 1);
  const int gbrow = (blockN + lr < N) ? (blockN + lr) : (N - 1);
  const bf16_t* aSrc = A  + (size_t)garow * K + lk;
  const bf16_t* bSrc = Bt + (size_t)gbrow * K + lk;

  auto stage_tile = [&](int b, int k0) {
#if USE_ASYNC_LDS
    async_copy32(aSrc + k0, &As[b][lr][lk]);
    async_copy32(bSrc + k0, &Bs[b][lr][lk]);
#else
    const uint4* sa = (const uint4*)(aSrc + k0);
    uint4 a0 = sa[0], a1 = sa[1];
    *(uint4*)&As[b][lr][lk]     = a0;
    *(uint4*)&As[b][lr][lk + 8] = a1;
    const uint4* sb = (const uint4*)(bSrc + k0);
    uint4 b0 = sb[0], b1 = sb[1];
    *(uint4*)&Bs[b][lr][lk]     = b0;
    *(uint4*)&Bs[b][lr][lk + 8] = b1;
#endif
  };

  stage_tile(0, 0);

  int buf = 0;
  for (int k0 = 0; k0 < K; k0 += GBK) {
    wait_async_all();     // this thread's DMA into As/Bs[buf] complete
    __syncthreads();      // everyone's DMA complete / previous reads retired

    if (k0 + GBK < K) {
      stage_tile(buf ^ 1, k0 + GBK);                 // overlap next fill
      __builtin_prefetch(aSrc + k0 + 2 * GBK, 0, 1); // pull k+2 toward L2
      __builtin_prefetch(bSrc + k0 + 2 * GBK, 0, 1);
    }

    const bf16_t* aBase = &As[buf][0][0];
    const bf16_t* bBase = &Bs[buf][0][0];
    v16bf afrag[4], bfrag[2];
    #pragma unroll
    for (int i = 0; i < 4; ++i)
      afrag[i] = load_frag_lds(aBase, wr * 64 + i * 16, lane);
    #pragma unroll
    for (int j = 0; j < 2; ++j)
      bfrag[j] = load_frag_lds(bBase, wc * 32 + j * 16, lane);

    #pragma unroll
    for (int i = 0; i < 4; ++i)
      #pragma unroll
      for (int j = 0; j < 2; ++j)
        acc[i][j] = __builtin_amdgcn_wmma_f32_16x16x32_bf16(
            false, afrag[i], false, bfrag[j], (short)0, acc[i][j], false, false);

    buf ^= 1;
  }

  // Writeback: VGPR r holds M = (lane<16 ? r : 8+r), N = lane & 15.
  #pragma unroll
  for (int i = 0; i < 4; ++i) {
    #pragma unroll
    for (int j = 0; j < 2; ++j) {
      const int m_base = blockM + wr * 64 + i * 16 + (lane >> 4) * 8;
      const int n      = blockN + wc * 32 + j * 16 + (lane & 15);
      if (n < N) {
        const float bn = bias ? bias[n] : 0.f;
        #pragma unroll
        for (int r = 0; r < 8; ++r) {
          const int m = m_base + r;
          if (m < M) {
            const float v = acc[i][j][r] + bn;
            if (Cf) Cf[(size_t)m * N + n] = v;
            if (Cb) Cb[(size_t)m * N + n] = (bf16_t)v;
          }
        }
      }
    }
  }
}

// ---------------------------------------------------------------------------
// Graph normalization: degrees (with self loop), dinv = rsqrt(deg).
// ---------------------------------------------------------------------------
__global__ void deg_init_kernel(float* __restrict__ deg, int n)
{
  int i = blockIdx.x * blockDim.x + threadIdx.x;
  if (i < n) deg[i] = 1.0f;   // self loop
}

__global__ void deg_edges_kernel(const int* __restrict__ ei, int nEdges, float* __restrict__ deg)
{
  int e = blockIdx.x * blockDim.x + threadIdx.x;
  if (e < nEdges) atomicAdd(&deg[ei[nEdges + e]], 1.0f);   // tgt row
}

__global__ void dinv_kernel(const float* __restrict__ deg, float* __restrict__ dinv, int n)
{
  int i = blockIdx.x * blockDim.x + threadIdx.x;
  if (i < n) dinv[i] = rsqrtf(deg[i]);
}

// agg[n][c] = bias[c] + dinv[n]^2 * m[n][c]   (self-loop term)
__global__ void agg_init_kernel(const float* __restrict__ m, const float* __restrict__ bias,
                                const float* __restrict__ dinv, int dim, int nNodes,
                                float* __restrict__ agg)
{
  const size_t total = (size_t)nNodes * dim;
  for (size_t i = (size_t)blockIdx.x * blockDim.x + threadIdx.x; i < total;
       i += (size_t)gridDim.x * blockDim.x) {
    const int n = (int)(i / dim);
    const int c = (int)(i % dim);
    const float d = dinv[n];
    agg[i] = bias[c] + d * d * m[i];
  }
}

// One block per edge; scatter coeff * m[src] into agg[tgt] (L2-resident atomics)
__global__ __launch_bounds__(256) void agg_edges_kernel(
    const float* __restrict__ m, const int* __restrict__ ei,
    const float* __restrict__ dinv, int dim, int nEdges, float* __restrict__ agg)
{
  const int e = blockIdx.x;
  if (e >= nEdges) return;
  const int s = ei[e];
  const int t = ei[nEdges + e];
  const float coeff = dinv[s] * dinv[t];
  const float* ms = m + (size_t)s * dim;
  float* at = agg + (size_t)t * dim;
  for (int c = threadIdx.x; c < dim; c += blockDim.x)
    atomicAdd(&at[c], coeff * ms[c]);
}

// ---------------------------------------------------------------------------
// BatchNorm over nodes (applied to leaky(agg)): one block per channel.
// ---------------------------------------------------------------------------
__global__ __launch_bounds__(256) void bn_stats_kernel(
    const float* __restrict__ agg, int nNodes, int dim,
    float* __restrict__ mean, float* __restrict__ var)
{
  const int c = blockIdx.x;
  __shared__ float s1[256];
  __shared__ float s2[256];
  float a = 0.f, b = 0.f;
  for (int n = threadIdx.x; n < nNodes; n += 256) {
    const float v = dleaky(agg[(size_t)n * dim + c]);
    a += v; b += v * v;
  }
  s1[threadIdx.x] = a; s2[threadIdx.x] = b;
  __syncthreads();
  for (int st = 128; st > 0; st >>= 1) {
    if (threadIdx.x < st) {
      s1[threadIdx.x] += s1[threadIdx.x + st];
      s2[threadIdx.x] += s2[threadIdx.x + st];
    }
    __syncthreads();
  }
  if (threadIdx.x == 0) {
    const float mu = s1[0] / (float)nNodes;
    mean[c] = mu;
    var[c]  = s2[0] / (float)nNodes - mu * mu;
  }
}

__global__ void bn_apply_kernel(const float* __restrict__ agg,
                                const float* __restrict__ mean, const float* __restrict__ var,
                                const float* __restrict__ g, const float* __restrict__ bt,
                                int dim, int nNodes,
                                bf16_t* __restrict__ outB, float* __restrict__ outF)
{
  const size_t total = (size_t)nNodes * dim;
  for (size_t i = (size_t)blockIdx.x * blockDim.x + threadIdx.x; i < total;
       i += (size_t)gridDim.x * blockDim.x) {
    const int c = (int)(i % dim);
    const float v = dleaky(agg[i]);
    const float y = (v - mean[c]) * rsqrtf(var[c] + BN_EPS) * g[c] + bt[c];
    outB[i] = (bf16_t)y;
    if (outF) outF[i] = y;
  }
}

// ---------------------------------------------------------------------------
// Graph pooling + MLP head (tiny, fp32).
// ---------------------------------------------------------------------------
__global__ void pool_init_kernel(float* __restrict__ pooled)
{
  int i = blockIdx.x * blockDim.x + threadIdx.x;
  if (i < N_GRAPHS * POOL_DIM) pooled[i] = 0.f;
}

__global__ __launch_bounds__(256) void pool_kernel(const float* __restrict__ h,
                                                   const int* __restrict__ batch,
                                                   float* __restrict__ pooled, int nNodes)
{
  const int n = blockIdx.x;
  if (n >= nNodes) return;
  const int grp = batch[n];
  const float* hr = h + (size_t)n * POOL_DIM;
  float* pr = pooled + (size_t)grp * POOL_DIM;
  for (int c = threadIdx.x; c < POOL_DIM; c += blockDim.x)
    atomicAdd(&pr[c], hr[c]);
}

__global__ void mlp_kernel(const float* __restrict__ A, const float* __restrict__ W,
                           const float* __restrict__ b, float* __restrict__ out,
                           int K, int N, int applyLeaky)
{
  const int m = blockIdx.x;
  const int n = threadIdx.x;
  if (n >= N) return;
  float a = b[n];
  const float* ar = A + (size_t)m * K;
  for (int k = 0; k < K; ++k) a += ar[k] * W[(size_t)k * N + n];
  if (applyLeaky) a = dleaky(a);
  out[(size_t)m * N + n] = a;
}

// ---------------------------------------------------------------------------
// Host launcher
// ---------------------------------------------------------------------------
extern "C" void kernel_launch(void* const* d_in, const int* in_sizes, int n_in,
                              void* d_out, int out_size, void* d_ws, size_t ws_size,
                              hipStream_t stream)
{
  (void)n_in; (void)out_size; (void)ws_size;

  const float* x        = (const float*)d_in[0];
  const int*   ei       = (const int*)d_in[1];
  const int*   batch    = (const int*)d_in[2];
  const float* conv1_w  = (const float*)d_in[3];
  const float* conv1_b  = (const float*)d_in[4];
  const float* conv2_w  = (const float*)d_in[5];
  const float* conv2_b  = (const float*)d_in[6];
  const float* conv3_w  = (const float*)d_in[7];
  const float* conv3_b  = (const float*)d_in[8];
  const float* proj_w   = (const float*)d_in[9];
  const float* proj_b   = (const float*)d_in[10];
  const float* gcn_w[4] = {(const float*)d_in[11], (const float*)d_in[12],
                           (const float*)d_in[13], (const float*)d_in[14]};
  const float* gcn_b[4] = {(const float*)d_in[15], (const float*)d_in[16],
                           (const float*)d_in[17], (const float*)d_in[18]};
  const float* bn_g[4]  = {(const float*)d_in[19], (const float*)d_in[20],
                           (const float*)d_in[21], (const float*)d_in[22]};
  const float* bn_b[4]  = {(const float*)d_in[23], (const float*)d_in[24],
                           (const float*)d_in[25], (const float*)d_in[26]};
  const float* mlp_w[3] = {(const float*)d_in[27], (const float*)d_in[28],
                           (const float*)d_in[29]};
  const float* mlp_b[3] = {(const float*)d_in[30], (const float*)d_in[31],
                           (const float*)d_in[32]};
  float* out = (float*)d_out;

  const int nNodes = in_sizes[2];         // 19456
  const int nEdges = in_sizes[1] / 2;     // 350000
  const int Kd[4] = {1280, 640, 512, 256};
  const int Nd[4] = {640, 512, 256, 256};

  // Workspace carve-out
  char* ws = (char*)d_ws;
  size_t off = 0;
  auto carve = [&](size_t bytes) -> char* {
    char* p = ws + off;
    off = (off + bytes + 255) & ~(size_t)255;
    return p;
  };
  bf16_t* wt_proj = (bf16_t*)carve((size_t)1280 * 768 * 2);
  bf16_t* wt_gcn[4];
  for (int l = 0; l < 4; ++l) wt_gcn[l] = (bf16_t*)carve((size_t)Nd[l] * Kd[l] * 2);
  bf16_t* h0b    = (bf16_t*)carve((size_t)nNodes * 768 * 2);
  bf16_t* h1b    = (bf16_t*)carve((size_t)nNodes * 1280 * 2);
  float*  m_f32  = (float*) carve((size_t)nNodes * 640 * 4);
  float*  agg    = (float*) carve((size_t)nNodes * 640 * 4);
  bf16_t* hb     = (bf16_t*)carve((size_t)nNodes * 640 * 2);
  float*  hlast  = (float*) carve((size_t)nNodes * 256 * 4);
  float*  deg    = (float*) carve((size_t)nNodes * 4);
  float*  dinv   = (float*) carve((size_t)nNodes * 4);
  float*  meanb  = (float*) carve((size_t)1280 * 4);
  float*  varb   = (float*) carve((size_t)1280 * 4);
  float*  pooled = (float*) carve((size_t)N_GRAPHS * POOL_DIM * 4);
  float*  t1     = (float*) carve((size_t)N_GRAPHS * 128 * 4);
  float*  t2     = (float*) carve((size_t)N_GRAPHS * 64 * 4);

  // 1) Weight conversion (fp32 KxN -> bf16 NxKpad)
  wt_convert_kernel<<<1024, 256, 0, stream>>>(proj_w, wt_proj, 750, 1280, 768);
  for (int l = 0; l < 4; ++l)
    wt_convert_kernel<<<1024, 256, 0, stream>>>(gcn_w[l], wt_gcn[l], Kd[l], Nd[l], Kd[l]);

  // 2) Conv stack -> h0 bf16 (nNodes x 768)
  conv_stack_kernel<<<nNodes, 256, 0, stream>>>(x, conv1_w, conv1_b, conv2_w, conv2_b,
                                                conv3_w, conv3_b, h0b);

  // 3) Degrees / dinv
  {
    int nb = (nNodes + 255) / 256;
    deg_init_kernel<<<nb, 256, 0, stream>>>(deg, nNodes);
    deg_edges_kernel<<<(nEdges + 255) / 256, 256, 0, stream>>>(ei, nEdges, deg);
    dinv_kernel<<<nb, 256, 0, stream>>>(deg, dinv, nNodes);
  }

  // 4) proj GEMM: h1 = h0 @ proj_w + proj_b   (bf16 out for next GEMM)
  {
    dim3 grid(1280 / GBN, (nNodes + GBM - 1) / GBM);
    gemm_bf16_kernel<<<grid, 256, 0, stream>>>(h0b, wt_proj, proj_b,
                                               nullptr, h1b, nNodes, 1280, 768);
  }

  // 5) GCN layers
  const bf16_t* Acur = h1b;
  for (int l = 0; l < 4; ++l) {
    const int K = Kd[l], N = Nd[l];
    // m = h @ W (fp32 out for gather/scatter)
    dim3 grid(N / GBN, (nNodes + GBM - 1) / GBM);
    gemm_bf16_kernel<<<grid, 256, 0, stream>>>(Acur, wt_gcn[l], nullptr,
                                               m_f32, nullptr, nNodes, N, K);
    // agg = bias + self-loop + edge scatter
    size_t total = (size_t)nNodes * N;
    int nbE = (int)((total + 255) / 256);
    agg_init_kernel<<<nbE, 256, 0, stream>>>(m_f32, gcn_b[l], dinv, N, nNodes, agg);
    agg_edges_kernel<<<nEdges, 256, 0, stream>>>(m_f32, ei, dinv, N, nEdges, agg);
    // leaky + batchnorm -> bf16 (and fp32 on last layer for pooling)
    bn_stats_kernel<<<N, 256, 0, stream>>>(agg, nNodes, N, meanb, varb);
    bn_apply_kernel<<<nbE, 256, 0, stream>>>(agg, meanb, varb, bn_g[l], bn_b[l],
                                             N, nNodes, hb, (l == 3) ? hlast : nullptr);
    Acur = hb;
  }

  // 6) Pool per graph
  pool_init_kernel<<<(N_GRAPHS * POOL_DIM + 255) / 256, 256, 0, stream>>>(pooled);
  pool_kernel<<<nNodes, 256, 0, stream>>>(hlast, batch, pooled, nNodes);

  // 7) MLP head: 256 -> 128 -> 64 -> 1
  mlp_kernel<<<N_GRAPHS, 128, 0, stream>>>(pooled, mlp_w[0], mlp_b[0], t1, 256, 128, 1);
  mlp_kernel<<<N_GRAPHS, 64, 0, stream>>>(t1, mlp_w[1], mlp_b[1], t2, 128, 64, 1);
  mlp_kernel<<<N_GRAPHS, 32, 0, stream>>>(t2, mlp_w[2], mlp_b[2], out, 64, 1, 0);
}
